// GAE_22385369547416
// MI455X (gfx1250) — matrix operand order repaired and checked by reading, not compiled, not run on hardware
//
#include <hip/hip_runtime.h>
#include <hip/hip_bf16.h>

typedef __attribute__((ext_vector_type(16))) __bf16 v16bf;
typedef __attribute__((ext_vector_type(8)))  __bf16 v8bf;
typedef __attribute__((ext_vector_type(8)))  float  v8f;

// ---------------- CSR build ----------------
__global__ __launch_bounds__(256) void k_zero_i(int* p, int n) {
    int i = blockIdx.x * 256 + threadIdx.x;
    if (i < n) p[i] = 0;
}

__global__ __launch_bounds__(256) void k_count(const int* __restrict__ dst, int* counts, int e) {
    int i = blockIdx.x * 256 + threadIdx.x;
    if (i < e) atomicAdd(&counts[dst[i]], 1);
}

// dinv[i] = rsqrt(in_degree + 1 self-loop)
__global__ __launch_bounds__(256) void k_dinv(const int* __restrict__ counts, float* dinv, int n) {
    int i = blockIdx.x * 256 + threadIdx.x;
    if (i < n) dinv[i] = rsqrtf((float)counts[i] + 1.0f);
}

// single-block chunked exclusive scan: start[0..n-1], start[n] = total
__global__ __launch_bounds__(1024) void k_scan(const int* __restrict__ counts, int* __restrict__ start, int n) {
    __shared__ int sm[1024];
    __shared__ int base;
    const int tid = threadIdx.x;
    if (tid == 0) base = 0;
    __syncthreads();
    for (int chunk = 0; chunk < n; chunk += 1024) {
        int i = chunk + tid;
        int v = (i < n) ? counts[i] : 0;
        sm[tid] = v;
        __syncthreads();
#pragma unroll
        for (int off = 1; off < 1024; off <<= 1) {
            int t = (tid >= off) ? sm[tid - off] : 0;
            __syncthreads();
            sm[tid] += t;
            __syncthreads();
        }
        if (i < n) start[i] = base + sm[tid] - v;   // exclusive
        __syncthreads();
        if (tid == 0) base += sm[1023];
        __syncthreads();
    }
    if (tid == 0) start[n] = base;
}

// place (src, norm) into per-dst buckets
__global__ __launch_bounds__(256) void k_fill(const int* __restrict__ src, const int* __restrict__ dst,
                                              const float* __restrict__ dinv, const int* __restrict__ start,
                                              int* cursor, int* __restrict__ csr_src, float* __restrict__ csr_w,
                                              int e) {
    int i = blockIdx.x * 256 + threadIdx.x;
    if (i >= e) return;
    int d = dst[i];
    int s = src[i];
    int p = start[d] + atomicAdd(&cursor[d], 1);
    csr_src[p] = s;
    csr_w[p]   = dinv[s] * dinv[d];
}

// ---------------- conversions ----------------
__global__ __launch_bounds__(256) void k_cvt_bf16(const float* __restrict__ in, __bf16* __restrict__ out, size_t n) {
    size_t i = (size_t)blockIdx.x * 256 + threadIdx.x;
    if (i < n) out[i] = (__bf16)in[i];
}

// W is [K][N] row-major; write Wt[n*K + k] (N-major, bf16)
__global__ __launch_bounds__(256) void k_cvt_wt(const float* __restrict__ W, __bf16* __restrict__ Wt, int K, int N) {
    int i = blockIdx.x * 256 + threadIdx.x;
    if (i >= K * N) return;
    int k = i / N, n = i % N;
    Wt[(size_t)n * K + k] = (__bf16)W[i];
}

// ---------------- bf16 WMMA GEMM: C[M,N] = A[M,128] x B (Bt is [N,128]) ----------------
template <int NT>  // NT = N/16
__global__ __launch_bounds__(256) void k_gemm_bf16(const __bf16* __restrict__ A,
                                                   const __bf16* __restrict__ Bt,
                                                   float* __restrict__ C, int M, int N) {
    const int K = 128;
    const int lane = threadIdx.x & 31;
    const int wave = threadIdx.x >> 5;
    const int l16  = lane & 15;
    const int kh   = lane >> 4;          // 0/1: K-half select
    const int m0   = (blockIdx.x * 8 + wave) * 16;

    int arow = m0 + l16;
    if (arow >= M) arow = M - 1;         // clamp: keep EXEC full for WMMA

    v8f acc[NT];
#pragma unroll
    for (int i = 0; i < NT; ++i) acc[i] = (v8f){0,0,0,0,0,0,0,0};

    const __bf16* aptr = A + (size_t)arow * K + kh * 16;
    const __bf16* bptr = Bt + (size_t)l16 * K + kh * 16;

#pragma unroll
    for (int ks = 0; ks < 4; ++ks) {
        const int bk = ks * 32;
        v8bf lo = *(const v8bf*)(aptr + bk);
        v8bf hi = *(const v8bf*)(aptr + bk + 8);
        v16bf afrag = __builtin_shufflevector(lo, hi, 0,1,2,3,4,5,6,7,8,9,10,11,12,13,14,15);
#pragma unroll
        for (int nt = 0; nt < NT; ++nt) {
            v16bf bfrag = *(const v16bf*)(bptr + (size_t)nt * 16 * K + bk);
            acc[nt] = __builtin_amdgcn_wmma_f32_16x16x32_bf16(
                false, afrag, false, bfrag, (short)0, acc[nt], false, false);
        }
    }

#pragma unroll
    for (int nt = 0; nt < NT; ++nt) {
#pragma unroll
        for (int r = 0; r < 8; ++r) {
            int row = m0 + r + kh * 8;
            if (row < M) C[(size_t)row * N + nt * 16 + l16] = acc[nt][r];
        }
    }
}

// ---------------- CSR aggregation, fused bias+ReLU (+bf16 cast) ----------------
// one wave per node; lane owns C/32 contiguous channels
template <int C, bool OUT_BF16>
__global__ __launch_bounds__(256) void k_aggregate(const float* __restrict__ y,
                                                   const float* __restrict__ dinv,
                                                   const int* __restrict__ start,
                                                   const int* __restrict__ csr_src,
                                                   const float* __restrict__ csr_w,
                                                   const float* __restrict__ bias,
                                                   void* __restrict__ outp, int n) {
    constexpr int VEC = C / 32;          // 4 (C=128) or 2 (C=64)
    const int node = (blockIdx.x * 256 + threadIdx.x) >> 5;
    const int lane = threadIdx.x & 31;
    if (node >= n) return;               // uniform per wave
    const int c0 = lane * VEC;

    float acc[VEC];
    {   // self-loop: norm = dinv^2
        float dv = dinv[node];
        const float* yv = y + (size_t)node * C + c0;
#pragma unroll
        for (int k = 0; k < VEC; ++k) acc[k] = yv[k] * dv * dv;
    }

    const int jbeg = start[node], jend = start[node + 1];
    for (int j = jbeg; j < jend; j += 32) {
        // cooperative edge-metadata load, broadcast via shuffle
        int   idx = j + lane;
        int   cl  = idx < jend ? idx : jend - 1;
        int   sv  = csr_src[cl];
        float wv  = csr_w[cl];
        int   cnt = jend - j; if (cnt > 32) cnt = 32;
        for (int kk = 0; kk < cnt; ++kk) {
            int   s = __shfl(sv, kk, 32);
            float w = __shfl(wv, kk, 32);
            const float* ys = y + (size_t)s * C + c0;
            if constexpr (VEC == 4) {
                float4 t = *(const float4*)ys;
                acc[0] = fmaf(t.x, w, acc[0]);
                acc[1] = fmaf(t.y, w, acc[1]);
                acc[2] = fmaf(t.z, w, acc[2]);
                acc[3] = fmaf(t.w, w, acc[3]);
            } else {
                float2 t = *(const float2*)ys;
                acc[0] = fmaf(t.x, w, acc[0]);
                acc[1] = fmaf(t.y, w, acc[1]);
            }
        }
    }

#pragma unroll
    for (int k = 0; k < VEC; ++k) {
        float v = acc[k] + bias[c0 + k];
        v = v > 0.f ? v : 0.f;
        if constexpr (OUT_BF16) ((__bf16*)outp)[(size_t)node * C + c0 + k] = (__bf16)v;
        else                    ((float*)outp)[(size_t)node * C + c0 + k] = v;
    }
}

static inline int blks(size_t total) { return (int)((total + 255) / 256); }

extern "C" void kernel_launch(void* const* d_in, const int* in_sizes, int n_in,
                              void* d_out, int out_size, void* d_ws, size_t ws_size,
                              hipStream_t stream) {
    const float* x   = (const float*)d_in[0];
    const int*   src = (const int*)d_in[1];
    const int*   dst = (const int*)d_in[2];
    const float* W1  = (const float*)d_in[3];
    const float* b1  = (const float*)d_in[4];
    const float* W2  = (const float*)d_in[5];
    const float* b2  = (const float*)d_in[6];

    const int IN = 128, HID = 128, OUT = 64;
    const int N = in_sizes[0] / IN;   // 100000
    const int E = in_sizes[1];        // 1600000
    (void)n_in; (void)out_size; (void)ws_size;

    // workspace carve-out (256B aligned)
    char* ws = (char*)d_ws;
    size_t off = 0;
    auto take = [&](size_t bytes) -> char* {
        char* p = ws + off;
        off += (bytes + 255) & ~(size_t)255;
        return p;
    };
    int*    counts  = (int*)take((size_t)N * 4);
    int*    startp  = (int*)take((size_t)(N + 1) * 4);
    int*    cursor  = (int*)take((size_t)N * 4);
    float*  dinv    = (float*)take((size_t)N * 4);
    int*    csr_src = (int*)take((size_t)E * 4);
    float*  csr_w   = (float*)take((size_t)E * 4);
    __bf16* xbf     = (__bf16*)take((size_t)N * IN * 2);   // reused as h_bf16
    __bf16* w1t     = (__bf16*)take((size_t)IN * HID * 2);
    __bf16* w2t     = (__bf16*)take((size_t)HID * OUT * 2);
    float*  y1      = (float*)take((size_t)N * HID * 4);   // reused as z0
    __bf16* hbf     = xbf;
    float*  z0      = y1;
    float*  out     = (float*)d_out;

    // 1. CSR build + normalization
    k_zero_i<<<blks(N), 256, 0, stream>>>(counts, N);
    k_count<<<blks(E), 256, 0, stream>>>(dst, counts, E);
    k_dinv<<<blks(N), 256, 0, stream>>>(counts, dinv, N);
    k_scan<<<1, 1024, 0, stream>>>(counts, startp, N);
    k_zero_i<<<blks(N), 256, 0, stream>>>(cursor, N);
    k_fill<<<blks(E), 256, 0, stream>>>(src, dst, dinv, startp, cursor, csr_src, csr_w, E);

    // 2. precision conversion / weight transpose
    k_cvt_bf16<<<blks((size_t)N * IN), 256, 0, stream>>>(x, xbf, (size_t)N * IN);
    k_cvt_wt<<<blks(IN * HID), 256, 0, stream>>>(W1, w1t, IN, HID);
    k_cvt_wt<<<blks(HID * OUT), 256, 0, stream>>>(W2, w2t, HID, OUT);

    // 3. layer 1: y1 = x @ W1 (WMMA), then CSR aggregate + bias + ReLU -> bf16 h
    k_gemm_bf16<8><<<(N + 127) / 128, 256, 0, stream>>>(xbf, w1t, y1, N, HID);
    k_aggregate<128, true><<<(N + 7) / 8, 256, 0, stream>>>(y1, dinv, startp, csr_src, csr_w, b1, hbf, N);

    // 4. layer 2: z0 = h @ W2 (WMMA), then CSR aggregate + bias + ReLU -> fp32 out
    k_gemm_bf16<4><<<(N + 127) / 128, 256, 0, stream>>>(hbf, w2t, z0, N, OUT);
    k_aggregate<64, false><<<(N + 7) / 8, 256, 0, stream>>>(z0, dinv, startp, csr_src, csr_w, b2, out, N);
}
